// APP_47888885350562
// MI455X (gfx1250) — matrix operand order repaired and checked by reading, not compiled, not run on hardware
//
#include <hip/hip_runtime.h>
#include <hip/hip_bf16.h>
#include <math.h>

// -------------------- types for CDNA5 WMMA --------------------
typedef __attribute__((ext_vector_type(16))) __bf16 v16bf;
typedef __attribute__((ext_vector_type(8)))  __bf16 v8bf;
typedef __attribute__((ext_vector_type(8)))  float  v8f;

#define SHUF16(lo, hi) __builtin_shufflevector(lo, hi, 0,1,2,3,4,5,6,7,8,9,10,11,12,13,14,15)

#define F_DIM 512
#define H_DIM 256
#define C_DIM 64
#define ALPHA  0.1f
#define BETA   0.9f

// Per CDNA5 ISA 7.12.2, with the canonical fragment-element ordering the
// 16-bit A fragment for lane (l16, half) is two contiguous bf16 runs:
//   elements 0..7  = K[ 8*half     .. 8*half+7  ]
//   elements 8..15 = K[16+8*half   .. 16+8*half+7]
// and the B fragment (K-major, i.e. column-major weights) uses the same two runs
//   at column = l16 + tile base.

// ==================== weight transpose+convert: W[k][n] -> Wt[n][k] bf16 ====
__global__ void transpose_bf16(const float* __restrict__ W, __bf16* __restrict__ Wt,
                               int K, int Ncol) {
    int i = blockIdx.x * blockDim.x + threadIdx.x;   // over Ncol*K, output-major
    if (i >= K * Ncol) return;
    int col = i / K, k = i - col * K;
    Wt[i] = (__bf16)W[(size_t)k * Ncol + col];
}

// ==================== fused MLP: h = relu(x W1 + b1) W2 + b2 ====================
// One block (8 waves) per 16-row strip. GEMM1: wave w owns column tiles 2w,2w+1.
// GEMM2: hidden (bf16 in LDS) x W2t; waves 0..3 own one 16-col tile each.
__global__ __launch_bounds__(256) void mlp_kernel(
    const float* __restrict__ x,   const __bf16* __restrict__ W1t,
    const float* __restrict__ b1,  const __bf16* __restrict__ W2t,
    const float* __restrict__ b2,  float* __restrict__ h,
    float* __restrict__ z0, int N)
{
    __shared__ __bf16 ldsA[16][40];          // x tile (16x32) as bf16, stride mult of 8
    __shared__ __bf16 ldsH[16][H_DIM + 8];   // hidden activations, stride 264 (mult of 8)

    const int tid  = threadIdx.x;
    const int wave = tid >> 5;
    const int lane = tid & 31;
    const int half = lane >> 4;
    const int l16  = lane & 15;
    const int row0 = blockIdx.x * 16;

    v8f acc0 = {}; v8f acc1 = {};
    const int c0 = wave * 32;                // column base of this wave's tiles
    const __bf16* pB0 = W1t + (size_t)(c0 + l16)      * F_DIM + 8 * half;
    const __bf16* pB1 = W1t + (size_t)(c0 + 16 + l16) * F_DIM + 8 * half;

    // staging assignment: each thread converts 2 consecutive x elements
    const int sr = tid >> 4;                 // staging row 0..15
    const int sc = (tid & 15) * 2;           // staging col 0,2,..,30
    int gr = row0 + sr; if (gr >= N) gr = N - 1;
    const float* px = x + (size_t)gr * F_DIM + sc;

    for (int kk = 0; kk < F_DIM; kk += 32) {
        ldsA[sr][sc]     = (__bf16)px[kk];
        ldsA[sr][sc + 1] = (__bf16)px[kk + 1];
        if (kk + 32 < F_DIM) __builtin_prefetch(px + kk + 32, 0, 1);  // global_prefetch_b8
        __syncthreads();

        v8bf alo = *(const v8bf*)&ldsA[l16][8 * half];        // ds_load_b128
        v8bf ahi = *(const v8bf*)&ldsA[l16][16 + 8 * half];   // ds_load_b128
        v16bf a  = SHUF16(alo, ahi);
        v16bf b0 = SHUF16(*(const v8bf*)(pB0 + kk), *(const v8bf*)(pB0 + kk + 16));                 // 2x global_load_b128
        v16bf b1v = SHUF16(*(const v8bf*)(pB1 + kk), *(const v8bf*)(pB1 + kk + 16));

        acc0 = __builtin_amdgcn_wmma_f32_16x16x32_bf16(false, a, false, b0,
                                                       (short)0, acc0, false, false);
        acc1 = __builtin_amdgcn_wmma_f32_16x16x32_bf16(false, a, false, b1v,
                                                       (short)0, acc1, false, false);
        __syncthreads();
    }

    // bias + relu, stash hidden to LDS as bf16 (C/D layout: row = r + 8*half)
    {
        const int col0 = c0 + l16, col1 = c0 + 16 + l16;
        const float bb0 = b1[col0], bb1 = b1[col1];
#pragma unroll
        for (int r = 0; r < 8; ++r) {
            int row = r + (half << 3);
            float v0 = acc0[r] + bb0; v0 = v0 > 0.f ? v0 : 0.f;
            float v1 = acc1[r] + bb1; v1 = v1 > 0.f ? v1 : 0.f;
            ldsH[row][col0] = (__bf16)v0;
            ldsH[row][col1] = (__bf16)v1;
        }
    }
    __syncthreads();

    // GEMM2: 16x256 (LDS, bf16) x W2t (64x256 col-major bf16)
    if (wave < 4) {
        v8f acc = {};
        const int col = wave * 16 + l16;
        const __bf16* pB = W2t + (size_t)col * H_DIM + 8 * half;
#pragma unroll
        for (int kk = 0; kk < H_DIM; kk += 32) {
            v8bf alo = *(const v8bf*)&ldsH[l16][kk + 8 * half];
            v8bf ahi = *(const v8bf*)&ldsH[l16][kk + 16 + 8 * half];
            v16bf a  = SHUF16(alo, ahi);
            v16bf b  = SHUF16(*(const v8bf*)(pB + kk), *(const v8bf*)(pB + kk + 16));
            acc = __builtin_amdgcn_wmma_f32_16x16x32_bf16(false, a, false, b,
                                                          (short)0, acc, false, false);
        }
        const float bb = b2[col];
#pragma unroll
        for (int r = 0; r < 8; ++r) {
            int row = row0 + r + (half << 3);
            if (row < N) {
                float v = acc[r] + bb;
                h [(size_t)row * C_DIM + col] = v;
                z0[(size_t)row * C_DIM + col] = v;   // APPNP starts from z = h
            }
        }
    }
}

// ==================== degree / normalization ====================
__global__ void deg_init(float* __restrict__ deg, int N) {
    int i = blockIdx.x * blockDim.x + threadIdx.x;
    if (i < N) deg[i] = 1.0f;                 // self loop
}
__global__ void deg_edges(const int* __restrict__ dst, float* __restrict__ deg, int E) {
    int e = blockIdx.x * blockDim.x + threadIdx.x;
    if (e < E) atomicAdd(&deg[dst[e]], 1.0f);
}
__global__ void dinv_kernel(const float* __restrict__ deg, float* __restrict__ dinv, int N) {
    int i = blockIdx.x * blockDim.x + threadIdx.x;
    if (i < N) dinv[i] = rsqrtf(deg[i]);
}

// ==================== APPNP hop ====================
// znext = alpha*h + beta*dinv[i]^2*z   (self-loop folded into init)
__global__ void hop_init(const float* __restrict__ h, const float* __restrict__ z,
                         const float* __restrict__ dinv, float* __restrict__ znext, int n) {
    int i = blockIdx.x * blockDim.x + threadIdx.x;
    if (i >= n) return;
    float di = dinv[i >> 6];
    znext[i] = ALPHA * h[i] + BETA * di * di * z[i];
}
// 16 threads per edge, 4 channels each (float4 gather + 4 f32 atomics)
__global__ __launch_bounds__(256) void hop_scatter(
    const int* __restrict__ src, const int* __restrict__ dst,
    const float* __restrict__ dinv, const float* __restrict__ z,
    float* __restrict__ znext, int E)
{
    long long t = (long long)blockIdx.x * blockDim.x + threadIdx.x;
    int e = (int)(t >> 4);
    if (e >= E) return;
    int sub = (int)(t & 15);
    int s = src[e], d = dst[e];
    float coef = BETA * dinv[s] * dinv[d];
    const float4 v = *(const float4*)(z + (size_t)s * C_DIM + sub * 4);
    float* p = znext + (size_t)d * C_DIM + sub * 4;
    atomicAdd(p + 0, coef * v.x);
    atomicAdd(p + 1, coef * v.y);
    atomicAdd(p + 2, coef * v.z);
    atomicAdd(p + 3, coef * v.w);
}

// ==================== outputs ====================
// log_softmax over rows: one wave per row, 2 channels per lane, shuffle reduce
__global__ __launch_bounds__(256) void logsoftmax_rows(const float* __restrict__ z,
                                                       float* __restrict__ out, int N) {
    int row = blockIdx.x * 8 + (threadIdx.x >> 5);
    if (row >= N) return;
    int lane = threadIdx.x & 31;
    float v0 = z[(size_t)row * C_DIM + lane];
    float v1 = z[(size_t)row * C_DIM + 32 + lane];
    float m = fmaxf(v0, v1);
    for (int off = 16; off > 0; off >>= 1) m = fmaxf(m, __shfl_xor(m, off, 32));
    float s = expf(v0 - m) + expf(v1 - m);
    for (int off = 16; off > 0; off >>= 1) s += __shfl_xor(s, off, 32);
    float ls = logf(s);
    out[(size_t)row * C_DIM + lane]      = v0 - m - ls;
    out[(size_t)row * C_DIM + 32 + lane] = v1 - m - ls;
}

// monotonic float<->uint encoding for atomicMax
__device__ __forceinline__ unsigned fenc(float f) {
    unsigned u = __float_as_uint(f);
    return u ^ ((u >> 31) ? 0xFFFFFFFFu : 0x80000000u);
}
__device__ __forceinline__ float fdec(unsigned u) {
    unsigned b = (u >> 31) ? (u ^ 0x80000000u) : (u ^ 0xFFFFFFFFu);
    return __uint_as_float(b);
}
__global__ void col_init(unsigned* __restrict__ cmax, float* __restrict__ csum) {
    int i = threadIdx.x;
    if (i < C_DIM) { cmax[i] = 0u; csum[i] = 0.f; }
}
// stride = gridDim*256 is a multiple of 64 -> each thread stays in one column
__global__ __launch_bounds__(256) void col_max(const float* __restrict__ z,
                                               unsigned* __restrict__ cmax, int n) {
    int tid = blockIdx.x * 256 + threadIdx.x;
    int stride = gridDim.x * 256;
    float m = -3.402823466e38f;
    for (int i = tid; i < n; i += stride) m = fmaxf(m, z[i]);
    atomicMax(&cmax[tid & 63], fenc(m));
}
__global__ __launch_bounds__(256) void col_sum(const float* __restrict__ z,
                                               const unsigned* __restrict__ cmax,
                                               float* __restrict__ csum, int n) {
    int tid = blockIdx.x * 256 + threadIdx.x;
    int stride = gridDim.x * 256;
    float m = fdec(cmax[tid & 63]);
    float s = 0.f;
    for (int i = tid; i < n; i += stride) s += expf(z[i] - m);
    atomicAdd(&csum[tid & 63], s);
}
__global__ void finalize(const float* __restrict__ z, const unsigned* __restrict__ cmax,
                         const float* __restrict__ csum, float* __restrict__ out1,
                         float* __restrict__ out2, int n) {
    int i = blockIdx.x * blockDim.x + threadIdx.x;
    if (i >= n) return;
    int col = i & 63;
    float zi = z[i];
    out1[i] = zi;
    out2[i] = expf(zi - fdec(cmax[col])) / csum[col];
}

// ==================== launch ====================
extern "C" void kernel_launch(void* const* d_in, const int* in_sizes, int n_in,
                              void* d_out, int out_size, void* d_ws, size_t ws_size,
                              hipStream_t stream) {
    const float* x  = (const float*)d_in[0];
    const int*   ei = (const int*)  d_in[1];   // [2][E]
    const float* W1 = (const float*)d_in[2];
    const float* b1 = (const float*)d_in[3];
    const float* W2 = (const float*)d_in[4];
    const float* b2 = (const float*)d_in[5];

    const int N = in_sizes[0] / F_DIM;
    const int E = in_sizes[1] / 2;
    const int* esrc = ei;
    const int* edst = ei + E;
    const size_t nc = (size_t)N * C_DIM;

    // workspace carve-up (~78.3 MB total); every sub-buffer offset stays 32B aligned
    float*    h     = (float*)d_ws;
    float*    zA    = h  + nc;
    float*    zB    = zA + nc;
    float*    deg   = zB + nc;
    float*    dinv  = deg + N;
    unsigned* cmax  = (unsigned*)(dinv + N);
    float*    csum  = (float*)(cmax + C_DIM);
    __bf16*   W1t   = (__bf16*)(csum + C_DIM);            // 256 KB, col-major bf16
    __bf16*   W2t   = W1t + (size_t)F_DIM * H_DIM;        // 32 KB,  col-major bf16

    // weight prep (tiny)
    transpose_bf16<<<(F_DIM * H_DIM + 255) / 256, 256, 0, stream>>>(W1, W1t, F_DIM, H_DIM);
    transpose_bf16<<<(H_DIM * C_DIM + 255) / 256, 256, 0, stream>>>(W2, W2t, H_DIM, C_DIM);

    // MLP (WMMA) -> h, z0
    mlp_kernel<<<(N + 15) / 16, 256, 0, stream>>>(x, W1t, b1, W2t, b2, h, zA, N);

    // GCN normalization
    deg_init   <<<(N + 255) / 256, 256, 0, stream>>>(deg, N);
    deg_edges  <<<(E + 255) / 256, 256, 0, stream>>>(edst, deg, E);
    dinv_kernel<<<(N + 255) / 256, 256, 0, stream>>>(deg, dinv, N);

    // 10 APPNP hops, ping-pong z buffers
    float* zc = zA; float* zn = zB;
    const long long st = (long long)E * 16;
    for (int hop = 0; hop < 10; ++hop) {
        hop_init   <<<(int)((nc + 255) / 256), 256, 0, stream>>>(h, zc, dinv, zn, (int)nc);
        hop_scatter<<<(int)((st + 255) / 256), 256, 0, stream>>>(esrc, edst, dinv, zc, zn, E);
        float* t = zc; zc = zn; zn = t;
    }

    // outputs: [log_softmax | z | softmax_axis0]
    float* out0 = (float*)d_out;
    float* out1 = out0 + nc;
    float* out2 = out0 + 2 * nc;
    logsoftmax_rows<<<(N + 7) / 8, 256, 0, stream>>>(zc, out0, N);
    col_init<<<1, 64, 0, stream>>>(cmax, csum);
    col_max <<<256, 256, 0, stream>>>(zc, cmax, (int)nc);
    col_sum <<<256, 256, 0, stream>>>(zc, cmax, csum, (int)nc);
    finalize<<<(int)((nc + 255) / 256), 256, 0, stream>>>(zc, cmax, csum, out1, out2, (int)nc);
}
